// MultiHeadPosAtt_25915832664789
// MI455X (gfx1250) — matrix operand order repaired and checked by reading, not compiled, not run on hardware
//
#include <hip/hip_runtime.h>
#include <cmath>
#include <cstdint>

typedef __attribute__((ext_vector_type(16))) _Float16 v16h;
typedef __attribute__((ext_vector_type(8)))  float    v8f;

#define B_    4
#define N_    4096
#define C_    128
#define H_    8
#define V_    16
#define COLS  64            // B_*V_
#define PI_F  3.14159265358979f

// ---------------------------------------------------------------------------
// Stage 1: per-head bandwidth t_h = tan(0.25*pi*(1+sin(r_h)))   (t_h > 0 a.s.)
// ---------------------------------------------------------------------------
__global__ void head_scale_kernel(const float* __restrict__ r,
                                  float* __restrict__ t) {
  int h = threadIdx.x;
  if (h < H_) t[h] = tanf(0.25f * PI_F * (1.0f + sinf(r[h])));
}

// ---------------------------------------------------------------------------
// Stage 2: per-row stats of dist. One workgroup per row.
//  - bitonic sort 4096 f32 in LDS
//  - p_n : linear-interp percentile (index locality/100*(N-1))
//  - dmin_n : row min (softmax max-shift, valid since t_h>0)
//  - inv_s[h][n] : 1 / sum_{d<=p} exp(-t_h*(d-dmin))   (order-independent sum)
// ---------------------------------------------------------------------------
__global__ __launch_bounds__(256)
void row_stats_kernel(const float* __restrict__ dist,
                      const float* __restrict__ t,
                      const int*   __restrict__ locality,
                      float* __restrict__ p_out,
                      float* __restrict__ dmin_out,
                      float* __restrict__ inv_s_out) {
  __shared__ float s[N_];
  __shared__ float red[256];
  const int n   = blockIdx.x;
  const int tid = threadIdx.x;
  const float* row = dist + (size_t)n * N_;
  for (int i = tid; i < N_; i += 256) s[i] = row[i];
  __syncthreads();

  // bitonic sort, ascending
  for (int k = 2; k <= N_; k <<= 1) {
    for (int j = k >> 1; j > 0; j >>= 1) {
      for (int i = tid; i < N_; i += 256) {
        int ixj = i ^ j;
        if (ixj > i) {
          float a = s[i], b = s[ixj];
          bool up = ((i & k) == 0);
          bool sw = up ? (a > b) : (a < b);
          if (sw) { s[i] = b; s[ixj] = a; }
        }
      }
      __syncthreads();
    }
  }

  const float q    = (float)(*locality) * 0.01f * (float)(N_ - 1);
  const int   lo   = (int)floorf(q);
  const float frac = q - (float)lo;
  const int   hi2  = (lo + 1 < N_) ? lo + 1 : N_ - 1;
  const float p    = s[lo] + frac * (s[hi2] - s[lo]);
  const float dmin = s[0];

  for (int h = 0; h < H_; ++h) {
    const float th = t[h];
    float part = 0.0f;
    for (int i = tid; i < N_; i += 256) {
      float d = s[i];
      if (d <= p) part += __expf(-th * (d - dmin));
    }
    red[tid] = part;
    __syncthreads();
    for (int w = 128; w > 0; w >>= 1) {
      if (tid < w) red[tid] += red[tid + w];
      __syncthreads();
    }
    if (tid == 0) inv_s_out[h * N_ + n] = 1.0f / red[0];
    __syncthreads();
  }
  if (tid == 0) { p_out[n] = p; dmin_out[n] = dmin; }
}

// ---------------------------------------------------------------------------
// Stage 3: value = einsum bnj,hjk->bhnk, stored f16 BLOCK-TRANSPOSED:
//   vmat[ ((h*(N/32) + j/32)*64 + (b*16+v))*32 + (j%32) ]
// so each GEMM K-tile (fixed h, 32 j's, all 64 cols) is one contiguous 4 KB
// block already in [col][k] order -> async-DMA straight into LDS, no shuffle.
// ---------------------------------------------------------------------------
__global__ __launch_bounds__(256)
void value_kernel(const float* __restrict__ inputs,
                  const float* __restrict__ weight,
                  _Float16* __restrict__ vmat) {
  const int tid = blockIdx.x * 256 + threadIdx.x;   // h*(N*64) + j*64 + col
  const int col = tid & (COLS - 1);
  const int j   = (tid >> 6) & (N_ - 1);
  const int h   = tid >> 18;
  const int b   = col >> 4;
  const int v   = col & 15;
  const float* in = inputs + ((size_t)b * N_ + j) * C_;
  const float* w  = weight + (size_t)h * C_ * V_ + v;
  float acc = 0.0f;
#pragma unroll 4
  for (int c = 0; c < C_; ++c) acc += in[c] * w[c * V_];
  const int jb = j >> 5, jin = j & 31;
  vmat[(((size_t)h * (N_ / 32) + jb) * COLS + col) * 32 + jin] = (_Float16)acc;
}

// ---------------------------------------------------------------------------
// Stage 4: out[b,n,h*16+v] = gelu( sum_j att[h,n,j] * value )
// att computed on the fly, f16 WMMA 16x16x32, fp32 accumulate.
// Double-buffered async global->LDS staging (GLOBAL_LOAD_ASYNC_TO_LDS_B128,
// ASYNCcnt): 6 DMA issues/thread/iter, s_wait_asynccnt 6 overlaps tile i+1's
// DMA with tile i's compute (async loads complete in order).
// Block = 128 threads (4 wave32). Block tile: 64 rows(n) x 64 cols, K = 4096.
// ---------------------------------------------------------------------------
#define DSTR 36   // lds_d row stride (f32): chunks stay 16B aligned, banks spread
#define BSTR 48   // lds_b col stride (f16): 96B => 32B-aligned v16h

__global__ __launch_bounds__(128)
void attn_gemm_kernel(const float* __restrict__ dist,
                      const _Float16* __restrict__ vmat,
                      const float* __restrict__ t,
                      const float* __restrict__ p_row,
                      const float* __restrict__ dmin_row,
                      const float* __restrict__ inv_s,
                      float* __restrict__ out) {
  __shared__ float    lds_d[2][64 * DSTR];   // [buf][row 0..63][k 0..31]
  __shared__ _Float16 lds_b[2][COLS * BSTR]; // [buf][col 0..63][k 0..31]

  const int h    = blockIdx.y;
  const int n0   = blockIdx.x * 64;
  const int tid  = threadIdx.x;
  const int wave = tid >> 5;
  const int lane = tid & 31;
  const int rlo  = lane & 15;       // A-frag row within wave strip / B-frag col
  const int hi   = lane >> 4;       // K-half selector
  const int myrow = wave * 16 + rlo;

  const float th = t[h];
  const float pr = p_row[n0 + myrow];
  const float dm = dmin_row[n0 + myrow];
  const float is = inv_s[h * N_ + n0 + myrow];

  // A-fragment K indices, contiguous order matching B:
  // element e, lanes 0-15: K = e; lanes 16-31: K = 16 + e
  int kidx[16];
#pragma unroll
  for (int e = 0; e < 16; ++e) {
    kidx[e] = hi * 16 + e;
  }

  v8f acc[4] = {};

  // ---- async tile issue: 4 dist chunks + 2 vmat chunks per thread (16B each)
  auto issue_tile = [&](int j0, int buf) {
#pragma unroll
    for (int q = 0; q < 4; ++q) {
      const int id  = tid + 128 * q;     // 0..511
      const int row = id >> 3;
      const int cf  = (id & 7) * 4;      // float offset within 32-wide row
      unsigned long long g =
          (unsigned long long)(uintptr_t)(dist + (size_t)(n0 + row) * N_ + j0 + cf);
      unsigned int l = (unsigned int)(uintptr_t)&lds_d[buf][row * DSTR + cf];
      asm volatile("global_load_async_to_lds_b128 %0, %1, off"
                   :: "v"(l), "v"(g) : "memory");
    }
    const _Float16* vb = vmat + (((size_t)h * (N_ / 32) + (j0 >> 5)) * COLS) * 32;
#pragma unroll
    for (int q = 0; q < 2; ++q) {
      const int c   = tid + 128 * q;     // 0..255 ; chunk = halves [c*8, c*8+8)
      const int col = c >> 2;
      const int kin = (c & 3) * 8;
      unsigned long long g = (unsigned long long)(uintptr_t)(vb + c * 8);
      unsigned int l = (unsigned int)(uintptr_t)&lds_b[buf][col * BSTR + kin];
      asm volatile("global_load_async_to_lds_b128 %0, %1, off"
                   :: "v"(l), "v"(g) : "memory");
    }
  };

  issue_tile(0, 0);
  const int NIT = N_ / 32;
  for (int it = 0; it < NIT; ++it) {
    const int buf = it & 1;
    if (it + 1 < NIT) {
      issue_tile((it + 1) * 32, buf ^ 1);               // prefetch next tile
      asm volatile("s_wait_asynccnt 0x6" ::: "memory"); // current tile landed
    } else {
      asm volatile("s_wait_asynccnt 0x0" ::: "memory");
    }
    __syncthreads();

    // A fragment: attention weights for this lane's row, f16
    v16h afrag;
#pragma unroll
    for (int e = 0; e < 16; ++e) {
      float d = lds_d[buf][myrow * DSTR + kidx[e]];
      float a = (d <= pr) ? __expf(-th * (d - dm)) * is : 0.0f;
      afrag[e] = (_Float16)a;
    }

    const int kb = hi * 16;
#pragma unroll
    for (int nt = 0; nt < 4; ++nt) {
      v16h bfrag = *(const v16h*)&lds_b[buf][(nt * 16 + rlo) * BSTR + kb];
      acc[nt] = __builtin_amdgcn_wmma_f32_16x16x32_f16(
          false, afrag, false, bfrag, (short)0, acc[nt], false, false);
    }
    __syncthreads();
  }

  // writeout: C/D layout -> VGPR i holds M = hi*8 + i, N = rlo; fuse exact gelu
  const float inv_sqrt2 = 0.70710678118654752f;
#pragma unroll
  for (int nt = 0; nt < 4; ++nt) {
    const int col = nt * 16 + rlo;
    const int b   = col >> 4;
    const int v   = col & 15;
#pragma unroll
    for (int i = 0; i < 8; ++i) {
      const int row = n0 + wave * 16 + hi * 8 + i;
      float x = acc[nt][i];
      float g = 0.5f * x * (1.0f + erff(x * inv_sqrt2));
      out[((size_t)b * N_ + row) * (H_ * V_) + h * V_ + v] = g;
    }
  }
}

// ---------------------------------------------------------------------------
// Host launcher
// ---------------------------------------------------------------------------
extern "C" void kernel_launch(void* const* d_in, const int* in_sizes, int n_in,
                              void* d_out, int out_size, void* d_ws, size_t ws_size,
                              hipStream_t stream) {
  const float* inputs   = (const float*)d_in[0];
  const float* dist     = (const float*)d_in[1];
  const float* r        = (const float*)d_in[2];
  const float* weight   = (const float*)d_in[3];
  const int*   locality = (const int*)d_in[4];

  float*    t     = (float*)d_ws;            // 8
  float*    p     = t + 8;                   // 4096
  float*    dmin  = p + N_;                  // 4096
  float*    inv_s = dmin + N_;               // 8*4096
  _Float16* vmat  = (_Float16*)(inv_s + H_ * N_);  // H*N*64 f16 (4 MB), 16B aligned
  float*    out   = (float*)d_out;

  head_scale_kernel<<<1, 8, 0, stream>>>(r, t);
  row_stats_kernel<<<N_, 256, 0, stream>>>(dist, t, locality, p, dmin, inv_s);
  value_kernel<<<(H_ * N_ * COLS) / 256, 256, 0, stream>>>(inputs, weight, vmat);
  attn_gemm_kernel<<<dim3(N_ / 64, H_), 128, 0, stream>>>(dist, vmat, t, p, dmin,
                                                          inv_s, out);
}
